// ProtoNetAlignQGPASR_88837103550585
// MI455X (gfx1250) — matrix operand order repaired and checked by reading, not compiled, not run on hardware
//
#include <hip/hip_runtime.h>

// ---------------------------------------------------------------------------
// Problem constants (from reference)
// ---------------------------------------------------------------------------
#define CDIM   320
#define NPTS   2048
#define MS     20480   // support rows = 2*5*2048
#define MQ     4096    // query rows   = 2*2048
#define KP     100
#define KPAD   112     // 7 * 16
#define NPROTO 300
#define NPPAD  304     // 19 * 16

typedef __attribute__((ext_vector_type(16))) _Float16 v16h;
typedef __attribute__((ext_vector_type(8)))  float    v8f;

union HFrag { v16h v; unsigned int u[8]; };

// A fragment: 16x32 f16, row r = lane&15, half h = lane>>4.
// halfs offset o(v) = h*16 + 2*v  -> dwords: h*8+v (contiguous K-range h*16..+15)
__device__ __forceinline__ v16h loadAfrag(const _Float16* __restrict__ rowAtKc, int h) {
    HFrag f;
    const unsigned int* p = (const unsigned int*)rowAtKc;
#pragma unroll
    for (int v = 0; v < 4; ++v) {
        f.u[v]     = p[h * 8 + v];
        f.u[4 + v] = p[h * 8 + 4 + v];
    }
    return f.v;
}

// B fragment: 32x16 f16. Lane holds column n = lane&15, K-range (lane>>4)*16 .. +15.
// colAtK points at B-source row n (row-major [n][c]) offset to kc + (lane>>4)*16.
__device__ __forceinline__ v16h loadBfrag(const _Float16* __restrict__ colAtK) {
    HFrag f;
    const unsigned int* p = (const unsigned int*)colAtK;
#pragma unroll
    for (int v = 0; v < 8; ++v) f.u[v] = p[v];
    return f.v;
}

// ---------------------------------------------------------------------------
// K1: per-point inverse L2 norm over C=320 (coalesced along pt)
// src layout: [slab][C][NPTS], id = slab*2048 + pt
// ---------------------------------------------------------------------------
__global__ __launch_bounds__(256) void k_norms(const float* __restrict__ src,
                                               float* __restrict__ inv, int total) {
    int id = blockIdx.x * 256 + threadIdx.x;
    if (id >= total) return;
    int slab = id >> 11, pt = id & 2047;
    const float* p = src + (size_t)slab * CDIM * NPTS + pt;
    float s = 0.f;
#pragma unroll 4
    for (int c = 0; c < CDIM; ++c) { float v = p[(size_t)c * NPTS]; s += v * v; }
    inv[id] = 1.0f / (sqrtf(s) + 1e-8f);
}

// ---------------------------------------------------------------------------
// K2: LDS-tiled transpose [C,NPTS] -> row-major [m, C]; writes f32 raw (opt)
//     and f16 normalized rows. grid (64, 10, nSlabs), block (32, 8)
// ---------------------------------------------------------------------------
__global__ __launch_bounds__(256) void k_transpose(const float* __restrict__ src,
                                                   const float* __restrict__ inv,
                                                   _Float16* __restrict__ fn,
                                                   float* __restrict__ raw) {
    __shared__ float tile[32][33];
    int slab = blockIdx.z;
    const float* s = src + (size_t)slab * CDIM * NPTS;
    int pt0 = blockIdx.x * 32, c0 = blockIdx.y * 32;
    int tx = threadIdx.x, ty = threadIdx.y;
#pragma unroll
    for (int i = 0; i < 4; ++i)
        tile[ty + i * 8][tx] = s[(size_t)(c0 + ty + i * 8) * NPTS + pt0 + tx];
    __syncthreads();
#pragma unroll
    for (int i = 0; i < 4; ++i) {
        int m = slab * NPTS + pt0 + ty + i * 8;
        int c = c0 + tx;
        float v = tile[tx][ty + i * 8];
        if (raw) raw[(size_t)m * CDIM + c] = v;
        fn[(size_t)m * CDIM + c] = (_Float16)(v * inv[m]);
    }
}

// ---------------------------------------------------------------------------
// K3: seed selection (first-100 masked points; fallback to unmasked) per group.
//     1 block, 3 waves (wave per group); wave32 ballot prefix scan.
// ---------------------------------------------------------------------------
__global__ __launch_bounds__(96) void k_seed_select(const int* __restrict__ sy,
                                                    int* __restrict__ seedIdx) {
    int g = threadIdx.x >> 5, lane = threadIdx.x & 31;
    int start = (g == 2) ? 10240 : 0;
    int len   = (g == 0) ? MS : 10240;
    int cnt = 0;
    for (int phase = 0; phase < 2 && cnt < KP; ++phase) {
        for (int base = 0; base < len && cnt < KP; base += 32) {
            int m = base + lane;
            int y = sy[start + m];
            bool p = (g == 0) ? (y == 0) : (y != 0);
            if (phase == 1) p = !p;
            unsigned bal = (unsigned)__ballot(p);
            int pre = __popc(bal & ((1u << lane) - 1u));
            if (p && (cnt + pre) < KP) seedIdx[g * KPAD + cnt + pre] = start + m;
            cnt += __popc(bal);
        }
    }
}

// ---------------------------------------------------------------------------
// K4: gather normalized seed rows into padded [3][112][320] f16 buffer
// ---------------------------------------------------------------------------
__global__ __launch_bounds__(64) void k_seed_gather(const _Float16* __restrict__ FnS,
                                                    const int* __restrict__ seedIdx,
                                                    _Float16* __restrict__ SeedsN) {
    int b = blockIdx.x;            // 0..299
    int g = b / KP, k = b % KP;
    int src = seedIdx[g * KPAD + k];
    const _Float16* s = FnS + (size_t)src * CDIM;
    _Float16* d = SeedsN + (size_t)(g * KPAD + k) * CDIM;
    for (int c = threadIdx.x; c < CDIM; c += 64) d[c] = s[c];
}

// ---------------------------------------------------------------------------
// K5: fused cosine-sim WMMA GEMM + per-row argmax over 100 seeds.
//     256 threads = 8 waves x 16 rows = 128 rows/block; 320 blocks cover
//     bg(160) + fg0(80) + fg1(80).
// ---------------------------------------------------------------------------
__global__ __launch_bounds__(256) void k_gemm_assign(const _Float16* __restrict__ FnS,
                                                     const _Float16* __restrict__ SeedsN,
                                                     int* __restrict__ assign) {
    int b = blockIdx.x;
    int g, localRow0, fnStart, asgBase;
    if (b < 160)      { g = 0; localRow0 = b * 128;         fnStart = 0;     asgBase = 0;     }
    else if (b < 240) { g = 1; localRow0 = (b - 160) * 128; fnStart = 0;     asgBase = MS;    }
    else              { g = 2; localRow0 = (b - 240) * 128; fnStart = 10240; asgBase = 30720; }
    const _Float16* seeds = SeedsN + (size_t)g * KPAD * CDIM;

    int wave = threadIdx.x >> 5, lane = threadIdx.x & 31;
    int row0 = localRow0 + wave * 16;
    int r = lane & 15, h = lane >> 4;
    const _Float16* arow = FnS + (size_t)(fnStart + row0 + r) * CDIM;
    int kHalf = h * 16;

    const v8f vzero = {};
    v8f acc[7];
#pragma unroll
    for (int t = 0; t < 7; ++t) acc[t] = vzero;

    for (int kc = 0; kc < CDIM; kc += 32) {
        v16h a = loadAfrag(arow + kc, h);
#pragma unroll
        for (int t = 0; t < 7; ++t) {
            v16h bf = loadBfrag(seeds + (size_t)(t * 16 + r) * CDIM + kc + kHalf);
            acc[t] = __builtin_amdgcn_wmma_f32_16x16x32_f16(
                false, a, false, bf, (short)0, acc[t], false, false);
        }
    }

    // D layout: lane holds column n = r, row m = v + 8*h.  argmax over n<100,
    // first-index tiebreak (matches jnp.argmax).
#pragma unroll
    for (int v = 0; v < 8; ++v) {
        float bv = -3.0e38f; int bi = 0;
#pragma unroll
        for (int t = 0; t < 7; ++t) {
            int n = t * 16 + r;
            float val = acc[t][v];
            if (n < KP && val > bv) { bv = val; bi = n; }
        }
#pragma unroll
        for (int sft = 1; sft <= 8; sft <<= 1) {
            float ov = __shfl_xor(bv, sft, 32);
            int   oi = __shfl_xor(bi, sft, 32);
            if (ov > bv || (ov == bv && oi < bi)) { bv = ov; bi = oi; }
        }
        if (r == 0) assign[asgBase + row0 + v + 8 * h] = bi;
    }
}

// ---------------------------------------------------------------------------
// K6: masked scatter-mean accumulation (one wave per point, coalesced rows,
//     f32 atomics into L2-resident num/den)
// ---------------------------------------------------------------------------
__global__ __launch_bounds__(256) void k_accum(const float* __restrict__ Fraw,
                                               const int* __restrict__ sy,
                                               const int* __restrict__ assign,
                                               float* __restrict__ num,
                                               float* __restrict__ den) {
    int e = blockIdx.x * 8 + (threadIdx.x >> 5);
    int lane = threadIdx.x & 31;
    int g, lm;
    if (e < MS)         { g = 0; lm = e; }
    else if (e < 30720) { g = 1; lm = e - MS; }
    else                { g = 2; lm = e - 30720; }
    int fnRow = (g == 2) ? (10240 + lm) : lm;
    int y = sy[fnRow];
    bool masked = (g == 0) ? (y == 0) : (y != 0);
    if (!masked) return;
    int p = g * KP + assign[e];
    const float* src = Fraw + (size_t)fnRow * CDIM;
    float* dst = num + (size_t)p * CDIM;
    for (int c = lane; c < CDIM; c += 32) atomicAdd(dst + c, src[c]);
    if (lane == 0) atomicAdd(den + p, 1.0f);
}

// ---------------------------------------------------------------------------
// K7: prototype normalization scale:
//     l2norm(num/den) = num / (|num| + 1e-8*(den+1e-8))
// ---------------------------------------------------------------------------
__global__ __launch_bounds__(320) void k_proto_scale(const float* __restrict__ num,
                                                     const float* __restrict__ den,
                                                     float* __restrict__ scale) {
    int p = threadIdx.x;
    if (p >= NPROTO) return;
    float s = 0.f;
    for (int c = 0; c < CDIM; ++c) { float v = num[(size_t)p * CDIM + c]; s += v * v; }
    scale[p] = 1.0f / (sqrtf(s) + 1e-8f * (den[p] + 1e-8f));
}

__global__ __launch_bounds__(256) void k_proto_write(const float* __restrict__ num,
                                                     const float* __restrict__ scale,
                                                     _Float16* __restrict__ PN) {
    int i = blockIdx.x * 256 + threadIdx.x;
    if (i >= NPPAD * CDIM) return;
    int p = i / CDIM;
    float v = (p < NPROTO) ? num[i] * scale[p] : 0.f;
    PN[i] = (_Float16)v;
}

// ---------------------------------------------------------------------------
// K8: fused query WMMA GEMM + per-class max (classes: n/100).
//     19 N-tiles chunked 5/5/5/4 to keep VGPRs modest. 32 blocks x 256 thr.
// ---------------------------------------------------------------------------
__global__ __launch_bounds__(256) void k_gemm_query(const _Float16* __restrict__ Qn,
                                                    const _Float16* __restrict__ PN,
                                                    float* __restrict__ pred,
                                                    float* __restrict__ out) {
    int wave = threadIdx.x >> 5, lane = threadIdx.x & 31;
    int row0 = blockIdx.x * 128 + wave * 16;
    int r = lane & 15, h = lane >> 4;
    const _Float16* arow = Qn + (size_t)(row0 + r) * CDIM;
    int kHalf = h * 16;
    const v8f vzero = {};

    float cm[8][3];
#pragma unroll
    for (int v = 0; v < 8; ++v)
#pragma unroll
        for (int c = 0; c < 3; ++c) cm[v][c] = -3.0e38f;

#pragma unroll
    for (int chunk = 0; chunk < 4; ++chunk) {
        const int t0 = chunk * 5;
        const int nt = (chunk == 3) ? 4 : 5;
        v8f acc[5];
#pragma unroll
        for (int t = 0; t < 5; ++t) acc[t] = vzero;
        for (int kc = 0; kc < CDIM; kc += 32) {
            v16h a = loadAfrag(arow + kc, h);
#pragma unroll
            for (int t = 0; t < 5; ++t) {
                if (t < nt) {
                    v16h bf = loadBfrag(PN + (size_t)((t0 + t) * 16 + r) * CDIM + kc + kHalf);
                    acc[t] = __builtin_amdgcn_wmma_f32_16x16x32_f16(
                        false, a, false, bf, (short)0, acc[t], false, false);
                }
            }
        }
#pragma unroll
        for (int t = 0; t < 5; ++t) {
            if (t < nt) {
                int n = (t0 + t) * 16 + r;
                if (n < NPROTO) {
                    int c = (n < 100) ? 0 : ((n < 200) ? 1 : 2);
#pragma unroll
                    for (int v = 0; v < 8; ++v) cm[v][c] = fmaxf(cm[v][c], acc[t][v]);
                }
            }
        }
    }

#pragma unroll
    for (int v = 0; v < 8; ++v) {
        int m = row0 + v + 8 * h;
#pragma unroll
        for (int c = 0; c < 3; ++c) {
            float mv = cm[v][c];
#pragma unroll
            for (int sft = 1; sft <= 8; sft <<= 1) mv = fmaxf(mv, __shfl_xor(mv, sft, 32));
            if (r == 0) {
                pred[(size_t)m * 3 + c] = mv;
                int q = m >> 11, pt = m & 2047;
                out[1 + (size_t)(q * 3 + c) * NPTS + pt] = mv;  // query_pred [2,3,2048]
            }
        }
    }
}

// ---------------------------------------------------------------------------
// K9: loss (2x mean CE via log-softmax) + per-point argmax/conf + conf mean
// ---------------------------------------------------------------------------
__global__ __launch_bounds__(256) void k_loss_conf(const float* __restrict__ pred,
                                                   const int* __restrict__ qy,
                                                   int* __restrict__ cls,
                                                   float* __restrict__ conf,
                                                   float* __restrict__ out,
                                                   float* __restrict__ scal) {
    __shared__ float sN[256], sC[256];
    int tid = threadIdx.x;
    float nll = 0.f, cs = 0.f;
    for (int m = tid; m < MQ; m += 256) {
        float p0 = pred[m * 3 + 0], p1 = pred[m * 3 + 1], p2 = pred[m * 3 + 2];
        float mx = fmaxf(p0, fmaxf(p1, p2));
        float lse = mx + logf(expf(p0 - mx) + expf(p1 - mx) + expf(p2 - mx));
        int y = qy[m];
        float py = (y == 0) ? p0 : ((y == 1) ? p1 : p2);
        nll += lse - py;
        int c = 0; float bv = p0;
        if (p1 > bv) { bv = p1; c = 1; }
        if (p2 > bv) { bv = p2; c = 2; }
        cls[m] = c; conf[m] = mx;
        cs += mx;
    }
    sN[tid] = nll; sC[tid] = cs;
    __syncthreads();
    for (int s = 128; s > 0; s >>= 1) {
        if (tid < s) { sN[tid] += sN[tid + s]; sC[tid] += sC[tid + s]; }
        __syncthreads();
    }
    if (tid == 0) { out[0] = 2.0f * sN[0] / (float)MQ; scal[0] = sC[0] / (float)MQ; }
}

__global__ __launch_bounds__(256) void k_final(const int* __restrict__ cls,
                                               const float* __restrict__ conf,
                                               const float* __restrict__ scal,
                                               float* __restrict__ out) {
    int m = blockIdx.x * 256 + threadIdx.x;
    if (m >= MQ) return;
    out[1 + 12288 + m] = (conf[m] > scal[0]) ? (float)cls[m] : -1.0f;
}

// ---------------------------------------------------------------------------
// Host-side launch (graph-capture safe: only memsetAsync + kernels on stream)
// ---------------------------------------------------------------------------
extern "C" void kernel_launch(void* const* d_in, const int* in_sizes, int n_in,
                              void* d_out, int out_size, void* d_ws, size_t ws_size,
                              hipStream_t stream) {
    (void)in_sizes; (void)n_in; (void)out_size; (void)ws_size;
    const float* sf = (const float*)d_in[0];  // [2,5,320,2048]
    const float* qf = (const float*)d_in[1];  // [2,320,2048]
    const int*   sy = (const int*)d_in[2];    // [2,5,2048]
    const int*   qy = (const int*)d_in[3];    // [2,2048]
    float* out = (float*)d_out;
    char* ws = (char*)d_ws;

    size_t o = 0;
    float*    Fraw    = (float*)(ws + o);    o += (size_t)MS * CDIM * 4;
    _Float16* FnS     = (_Float16*)(ws + o); o += (size_t)MS * CDIM * 2;
    _Float16* Qn      = (_Float16*)(ws + o); o += (size_t)MQ * CDIM * 2;
    float*    invS    = (float*)(ws + o);    o += (size_t)MS * 4;
    float*    invQ    = (float*)(ws + o);    o += (size_t)MQ * 4;
    int*      seedIdx = (int*)(ws + o);      o += (size_t)3 * KPAD * 4;
    _Float16* SeedsN  = (_Float16*)(ws + o); o += (size_t)3 * KPAD * CDIM * 2;
    int*      assign  = (int*)(ws + o);      o += (size_t)40960 * 4;
    float*    num     = (float*)(ws + o);    o += (size_t)NPROTO * CDIM * 4;
    float*    den     = (float*)(ws + o);    o += (size_t)NPPAD * 4;
    float*    scale   = (float*)(ws + o);    o += (size_t)NPPAD * 4;
    _Float16* PN      = (_Float16*)(ws + o); o += (size_t)NPPAD * CDIM * 2;
    float*    pred    = (float*)(ws + o);    o += (size_t)MQ * 3 * 4;
    int*      cls     = (int*)(ws + o);      o += (size_t)MQ * 4;
    float*    conf    = (float*)(ws + o);    o += (size_t)MQ * 4;
    float*    scal    = (float*)(ws + o);    o += 256;

    // zero-init accumulators + padded seed rows (num/den are contiguous)
    hipMemsetAsync(SeedsN, 0, (size_t)3 * KPAD * CDIM * 2, stream);
    hipMemsetAsync(num, 0, (size_t)NPROTO * CDIM * 4 + (size_t)NPPAD * 4, stream);

    k_norms<<<MS / 256, 256, 0, stream>>>(sf, invS, MS);
    k_norms<<<MQ / 256, 256, 0, stream>>>(qf, invQ, MQ);
    k_transpose<<<dim3(64, 10, 10), dim3(32, 8), 0, stream>>>(sf, invS, FnS, Fraw);
    k_transpose<<<dim3(64, 10, 2),  dim3(32, 8), 0, stream>>>(qf, invQ, Qn, nullptr);
    k_seed_select<<<1, 96, 0, stream>>>(sy, seedIdx);
    k_seed_gather<<<300, 64, 0, stream>>>(FnS, seedIdx, SeedsN);
    k_gemm_assign<<<320, 256, 0, stream>>>(FnS, SeedsN, assign);
    k_accum<<<5120, 256, 0, stream>>>(Fraw, sy, assign, num, den);
    k_proto_scale<<<1, 320, 0, stream>>>(num, den, scale);
    k_proto_write<<<(NPPAD * CDIM + 255) / 256, 256, 0, stream>>>(num, scale, PN);
    k_gemm_query<<<MQ / 128, 256, 0, stream>>>(Qn, PN, pred, out);
    k_loss_conf<<<1, 256, 0, stream>>>(pred, qy, cls, conf, out, scal);
    k_final<<<MQ / 256, 256, 0, stream>>>(cls, conf, scal, out);
}
